// MoE_55405078119405
// MI455X (gfx1250) — matrix operand order repaired and checked by reading, not compiled, not run on hardware
//
#include <hip/hip_runtime.h>
#include <hip/hip_fp16.h>

// ---------------------------------------------------------------------------
// MoE forward for MI455X (gfx1250, wave32, WMMA).
// Top-2 routed expert compute, 64-row tiles, v_wmma_f32_16x16x32_f16.
// ---------------------------------------------------------------------------

typedef __attribute__((ext_vector_type(16))) _Float16 v16h;
typedef __attribute__((ext_vector_type(8)))  float    v8f;

#define T_TOK   8192            // B*S tokens
#define DDIM    512
#define HDIM    512
#define NEXP    8
#define NPAIR   (T_TOK * 2)     // 16384 token-expert pairs
#define PADCAP  (NPAIR + NEXP * 64)  // list capacity incl. per-expert padding
#define TM      64              // rows per GEMM block (4 WMMA m-tiles)
#define KSTEP   32              // K per WMMA f16
#define ASTR    40              // LDS A slab row stride in halves (32 + 8 pad)
#define BSTR    40              // LDS B slab row stride in halves (32 + 8 pad)
#define MAXT64  (NPAIR / TM)    // 256 tiles worst case per expert

__device__ __forceinline__ unsigned pack2h(float lo, float hi) {
    union { _Float16 h[2]; unsigned u; } p;
    p.h[0] = (_Float16)lo; p.h[1] = (_Float16)hi;
    return p.u;
}

// ---------------------------------------------------------------------------
// Kernel 0: zero output + per-expert counters (deterministic each launch)
// ---------------------------------------------------------------------------
__global__ void k_zero(float* __restrict__ out, int* __restrict__ counts) {
    const int i = blockIdx.x * 256 + threadIdx.x;
    float4 z; z.x = 0.f; z.y = 0.f; z.z = 0.f; z.w = 0.f;
    ((float4*)out)[i] = z;
    if (blockIdx.x == 0 && threadIdx.x < NEXP) counts[threadIdx.x] = 0;
}

// ---------------------------------------------------------------------------
// Kernel 1: gating. One wave per token: logits = x . Wg, top-2 + softmax.
// ---------------------------------------------------------------------------
__global__ void __launch_bounds__(256) k_gate(
    const float* __restrict__ x, const float* __restrict__ Wg,
    int* __restrict__ topi, float* __restrict__ gatew, int* __restrict__ counts)
{
    const int t    = blockIdx.x * 8 + (threadIdx.x >> 5);
    const int lane = threadIdx.x & 31;
    float acc[NEXP];
#pragma unroll
    for (int e = 0; e < NEXP; ++e) acc[e] = 0.f;

    const float* xr = x + (size_t)t * DDIM;
#pragma unroll 4
    for (int i = 0; i < DDIM / 32; ++i) {
        const int d = i * 32 + lane;
        const float xv = xr[d];
        const float4* wr = (const float4*)(Wg + d * NEXP);
        const float4 w0 = wr[0], w1 = wr[1];
        acc[0] += xv * w0.x; acc[1] += xv * w0.y;
        acc[2] += xv * w0.z; acc[3] += xv * w0.w;
        acc[4] += xv * w1.x; acc[5] += xv * w1.y;
        acc[6] += xv * w1.z; acc[7] += xv * w1.w;
    }
#pragma unroll
    for (int off = 16; off > 0; off >>= 1)
#pragma unroll
        for (int e = 0; e < NEXP; ++e)
            acc[e] += __shfl_xor(acc[e], off, 32);

    if (lane == 0) {
        float v0 = -3.4e38f, v1 = -3.4e38f; int i0 = 0, i1 = 0;
#pragma unroll
        for (int e = 0; e < NEXP; ++e) {
            const float v = acc[e];
            if (v > v0) { v1 = v0; i1 = i0; v0 = v; i0 = e; }  // strict >: lax.top_k tie-break
            else if (v > v1) { v1 = v; i1 = e; }
        }
        const float e1 = __expf(v1 - v0);                      // v0 >= v1
        const float inv = 1.f / (1.f + e1);
        topi[t * 2] = i0;  topi[t * 2 + 1] = i1;
        gatew[t * 2] = inv; gatew[t * 2 + 1] = e1 * inv;
        atomicAdd(&counts[i0], 1);
        atomicAdd(&counts[i1], 1);
    }
}

// ---------------------------------------------------------------------------
// Kernel 2: exclusive scan over 8 expert counts, segments padded to 64 rows
// ---------------------------------------------------------------------------
__global__ void k_scan(const int* __restrict__ counts,
                       int* __restrict__ offs, int* __restrict__ cursor) {
    if (threadIdx.x == 0) {
        int o = 0;
        for (int e = 0; e < NEXP; ++e) {
            offs[e] = o; cursor[e] = o;
            o += (counts[e] + (TM - 1)) & ~(TM - 1);
        }
    }
}

// ---------------------------------------------------------------------------
// Kernel 3a: scatter tokens into compact per-expert lists
// ---------------------------------------------------------------------------
__global__ void k_scatter(const int* __restrict__ topi, const float* __restrict__ gatew,
                          int* __restrict__ cursor,
                          int* __restrict__ tok_list, float* __restrict__ w_list) {
    const int t = blockIdx.x * 256 + threadIdx.x;
#pragma unroll
    for (int k = 0; k < 2; ++k) {
        const int e = topi[t * 2 + k];
        const int p = atomicAdd(&cursor[e], 1);
        tok_list[p] = t;
        w_list[p]   = gatew[t * 2 + k];
    }
}

// ---------------------------------------------------------------------------
// Kernel 3b: fill per-expert padding with (token 0, weight 0) dummies.
// Padding rows compute finite values and contribute 0 via w=0.
// ---------------------------------------------------------------------------
__global__ void k_pad(const int* __restrict__ counts, const int* __restrict__ offs,
                      int* __restrict__ tok_list, float* __restrict__ w_list) {
    const int e = blockIdx.x;
    const int c = counts[e];
    const int pad = (c + (TM - 1)) & ~(TM - 1);
    const int i = c + threadIdx.x;
    if (i < pad) { tok_list[offs[e] + i] = 0; w_list[offs[e] + i] = 0.f; }
}

// ---------------------------------------------------------------------------
// WMMA fragment builds from padded LDS slabs (ISA 7.12.2 layouts).
// A 16x32 f16: lane m=lane&15, elem e -> K=(e>>3)*16 + hi*8 + (e&7)
// B 32x16 f16: lane n=lane&15, elem e -> K=(e>>3)*16 + hi*8 + (e&7)
// ---------------------------------------------------------------------------
__device__ __forceinline__ v16h load_a_frag(const _Float16* Al, int row, int hi) {
    v16h a;
    const uint4* p = (const uint4*)(Al + row * ASTR + hi * 8);
    ((uint4*)&a)[0] = p[0];     // K = hi*8 + 0..7
    ((uint4*)&a)[1] = p[2];     // K = 16 + hi*8 + 0..7
    return a;
}
__device__ __forceinline__ v16h load_b_frag(const _Float16* Bl, int n, int hi) {
    v16h b;
    const uint4* p = (const uint4*)(Bl + n * BSTR + hi * 8);
    ((uint4*)&b)[0] = p[0];
    ((uint4*)&b)[1] = p[2];
    return b;
}
// Stage one 32-K slab of W[e] (f32, K-major rows of 512) transposed into
// Bl[n][kk] f16, packing K-pairs into b32 LDS stores.
__device__ __forceinline__ void stage_b(_Float16* Bl, const float* __restrict__ Wp, int tid) {
#pragma unroll
    for (int i = 0; i < 8; ++i) {                 // 2048 items / 256 threads
        const int gid = i * 256 + tid;
        const int kp  = gid >> 7;                 // K-pair 0..15
        const int n4  = (gid & 127) << 2;         // 0..508
        const float* r0 = Wp + (size_t)(2 * kp) * DDIM + n4;
        const float4 a = *(const float4*)r0;
        const float4 b = *(const float4*)(r0 + DDIM);
        unsigned* q0 = (unsigned*)(Bl + (n4 + 0) * BSTR + 2 * kp);
        unsigned* q1 = (unsigned*)(Bl + (n4 + 1) * BSTR + 2 * kp);
        unsigned* q2 = (unsigned*)(Bl + (n4 + 2) * BSTR + 2 * kp);
        unsigned* q3 = (unsigned*)(Bl + (n4 + 3) * BSTR + 2 * kp);
        *q0 = pack2h(a.x, b.x); *q1 = pack2h(a.y, b.y);
        *q2 = pack2h(a.z, b.z); *q3 = pack2h(a.w, b.w);
    }
}

// ---------------------------------------------------------------------------
// Kernel 4: fc1 + ReLU. Block = (expert, 64 gathered rows) x N=512.
// 8 waves in 2x4 (M x N); 16 x v_wmma_f32_16x16x32_f16 per K-step.
// ---------------------------------------------------------------------------
__global__ void __launch_bounds__(256) k_fc1(
    const float* __restrict__ x, const float* __restrict__ W1, const float* __restrict__ b1,
    const int* __restrict__ counts, const int* __restrict__ offs,
    const int* __restrict__ tok_list, _Float16* __restrict__ hbuf)
{
    const int e = blockIdx.y, tile = blockIdx.x;
    const int cnt_pad = (counts[e] + (TM - 1)) & ~(TM - 1);
    if (tile * TM >= cnt_pad) return;
    const int pos0 = offs[e] + tile * TM;

    __shared__ alignas(16) _Float16 Al[TM * ASTR];      //  5 KB
    __shared__ alignas(16) _Float16 Bl[DDIM * BSTR];    // 40 KB

    const int tid  = threadIdx.x;
    const int lane = tid & 31, wave = tid >> 5;
    const int m = lane & 15, hi = lane >> 4;
    const int wave_m = wave >> 2, wave_n = wave & 3;    // 2 x 4 wave grid

    // per-thread A-stage assignment: row = tid>>2, 8 cols at (tid&3)*8
    const int arow = tid >> 2, acol = (tid & 3) * 8;
    const int tok  = tok_list[pos0 + arow];
    const float* xrow = x + (size_t)tok * DDIM + acol;
    _Float16* adst = Al + arow * ASTR + acol;

    v8f acc[2][8] = {};
    const float* We = W1 + (size_t)e * DDIM * HDIM;
    for (int kb = 0; kb < DDIM / KSTEP; ++kb) {
        __syncthreads();
        {   // stage A slab [64][32] f16
            const float4 v0 = ((const float4*)(xrow + kb * KSTEP))[0];
            const float4 v1 = ((const float4*)(xrow + kb * KSTEP))[1];
            uint4 pk;
            pk.x = pack2h(v0.x, v0.y); pk.y = pack2h(v0.z, v0.w);
            pk.z = pack2h(v1.x, v1.y); pk.w = pack2h(v1.z, v1.w);
            *(uint4*)adst = pk;
        }
        stage_b(Bl, We + (size_t)kb * KSTEP * HDIM, tid);
        if (kb + 1 < DDIM / KSTEP) {   // prefetch next W slab -> global_prefetch_b8
            const char* nxt = (const char*)(We + (size_t)(kb + 1) * KSTEP * HDIM);
            __builtin_prefetch(nxt + tid * 256, 0, 0);
            __builtin_prefetch(nxt + tid * 256 + 128, 0, 0);
        }
        __syncthreads();

        v16h a0 = load_a_frag(Al, wave_m * 32 + m, hi);
        v16h a1 = load_a_frag(Al, wave_m * 32 + 16 + m, hi);
#pragma unroll
        for (int jn = 0; jn < 8; ++jn) {
            const v16h b = load_b_frag(Bl, wave_n * 128 + jn * 16 + m, hi);
            acc[0][jn] = __builtin_amdgcn_wmma_f32_16x16x32_f16(
                false, a0, false, b, (short)0, acc[0][jn], false, false);
            acc[1][jn] = __builtin_amdgcn_wmma_f32_16x16x32_f16(
                false, a1, false, b, (short)0, acc[1][jn], false, false);
        }
    }

    // epilogue: + b1, ReLU, store h (f16); tiles always full (padded lists)
    const float* b1e = b1 + e * HDIM;
#pragma unroll
    for (int jn = 0; jn < 8; ++jn) {
        const int n = wave_n * 128 + jn * 16 + m;
        const float bias = b1e[n];
#pragma unroll
        for (int im = 0; im < 2; ++im) {
            const int rowbase = pos0 + wave_m * 32 + im * 16 + hi * 8;
#pragma unroll
            for (int r = 0; r < 8; ++r) {
                float v = acc[im][jn][r] + bias;
                v = v > 0.f ? v : 0.f;
                hbuf[(size_t)(rowbase + r) * HDIM + n] = (_Float16)v;
            }
        }
    }
}

// ---------------------------------------------------------------------------
// Kernel 5: fc2 + bias + LayerNorm + weighted scatter-add into out.
// LN runs in 4 rounds of 16 rows through LDS reused from the B slab.
// ---------------------------------------------------------------------------
__global__ void __launch_bounds__(256) k_fc2(
    const _Float16* __restrict__ hbuf, const float* __restrict__ W2, const float* __restrict__ b2,
    const float* __restrict__ gamma, const float* __restrict__ beta,
    const int* __restrict__ counts, const int* __restrict__ offs,
    const int* __restrict__ tok_list, const float* __restrict__ w_list,
    float* __restrict__ out)
{
    const int e = blockIdx.y, tile = blockIdx.x;
    const int cnt_pad = (counts[e] + (TM - 1)) & ~(TM - 1);
    if (tile * TM >= cnt_pad) return;
    const int pos0 = offs[e] + tile * TM;

    __shared__ alignas(16) _Float16 Al[TM * ASTR];
    __shared__ alignas(16) _Float16 Bl[DDIM * BSTR];    // reused as y[16][512] f32
    __shared__ float redS[16][16];
    __shared__ float redQ[16][16];

    const int tid  = threadIdx.x;
    const int lane = tid & 31, wave = tid >> 5;
    const int m = lane & 15, hi = lane >> 4;
    const int wave_m = wave >> 2, wave_n = wave & 3;

    const int arow = tid >> 2, acol = (tid & 3) * 8;
    const _Float16* hrow = hbuf + (size_t)(pos0 + arow) * HDIM + acol;
    _Float16* adst = Al + arow * ASTR + acol;

    v8f acc[2][8] = {};
    const float* We = W2 + (size_t)e * HDIM * DDIM;
    for (int kb = 0; kb < HDIM / KSTEP; ++kb) {
        __syncthreads();
        *(uint4*)adst = *(const uint4*)(hrow + kb * KSTEP);   // A slab (already f16)
        stage_b(Bl, We + (size_t)kb * KSTEP * DDIM, tid);
        if (kb + 1 < HDIM / KSTEP) {
            const char* nxt = (const char*)(We + (size_t)(kb + 1) * KSTEP * DDIM);
            __builtin_prefetch(nxt + tid * 256, 0, 0);
            __builtin_prefetch(nxt + tid * 256 + 128, 0, 0);
        }
        __syncthreads();

        v16h a0 = load_a_frag(Al, wave_m * 32 + m, hi);
        v16h a1 = load_a_frag(Al, wave_m * 32 + 16 + m, hi);
#pragma unroll
        for (int jn = 0; jn < 8; ++jn) {
            const v16h b = load_b_frag(Bl, wave_n * 128 + jn * 16 + m, hi);
            acc[0][jn] = __builtin_amdgcn_wmma_f32_16x16x32_f16(
                false, a0, false, b, (short)0, acc[0][jn], false, false);
            acc[1][jn] = __builtin_amdgcn_wmma_f32_16x16x32_f16(
                false, a1, false, b, (short)0, acc[1][jn], false, false);
        }
    }

    __syncthreads();                         // all waves done reading Bl
    float* yl = (float*)Bl;                  // 16 x 512 f32 = 32 KB <= 40 KB
    const float* b2e = b2 + e * DDIM;
    const float* ge  = gamma + e * DDIM;
    const float* be  = beta  + e * DDIM;
    const int row = tid >> 4, sub = tid & 15;

    for (int t = 0; t < 4; ++t) {            // rounds of 16 rows
        if (wave_m == (t >> 1)) {
            const int im = t & 1;
#pragma unroll
            for (int jn = 0; jn < 8; ++jn) {
                const int n = wave_n * 128 + jn * 16 + m;
                const float bias = b2e[n];
#pragma unroll
                for (int r = 0; r < 8; ++r)
                    yl[(r + hi * 8) * DDIM + n] = acc[im][jn][r] + bias;
            }
        }
        __syncthreads();

        // LayerNorm over D=512: 16 threads per row x 32 cols each
        const float* yr = yl + row * DDIM;
        float s = 0.f, q = 0.f;
#pragma unroll
        for (int c = sub * 32; c < sub * 32 + 32; ++c) { const float v = yr[c]; s += v; q += v * v; }
        redS[row][sub] = s; redQ[row][sub] = q;
        __syncthreads();
        float S = 0.f, Q = 0.f;
#pragma unroll
        for (int i = 0; i < 16; ++i) { S += redS[row][i]; Q += redQ[row][i]; }
        const float mu   = S * (1.f / (float)DDIM);
        const float var  = Q * (1.f / (float)DDIM) - mu * mu;
        const float rstd = rsqrtf(var + 1e-5f);

        const int p = pos0 + t * 16 + row;
        const int tok = tok_list[p];
        const float w = w_list[p];           // padding rows have w == 0
        float* op = out + (size_t)tok * DDIM;
#pragma unroll
        for (int c = sub * 32; c < sub * 32 + 32; ++c) {
            const float v = (yr[c] - mu) * rstd * ge[c] + be[c];
            unsafeAtomicAdd(op + c, w * v);  // exactly 2 (+zero) contribs: order-invariant
        }
        __syncthreads();                     // yl reused next round
    }
}

// ---------------------------------------------------------------------------
// Host launcher
// ---------------------------------------------------------------------------
extern "C" void kernel_launch(void* const* d_in, const int* in_sizes, int n_in,
                              void* d_out, int out_size, void* d_ws, size_t ws_size,
                              hipStream_t stream) {
    const float* x     = (const float*)d_in[0];   // [T, 512]
    const float* Wg    = (const float*)d_in[1];   // [512, 8]
    const float* W1    = (const float*)d_in[2];   // [8, 512, 512]
    const float* b1    = (const float*)d_in[3];   // [8, 512]
    const float* W2    = (const float*)d_in[4];   // [8, 512, 512]
    const float* b2    = (const float*)d_in[5];   // [8, 512]
    const float* gamma = (const float*)d_in[6];   // [8, 512]
    const float* beta  = (const float*)d_in[7];   // [8, 512]
    float* out = (float*)d_out;                   // [T, 512]

    // workspace carve-up (~17.6 MB)
    char* ws = (char*)d_ws;
    int*      counts = (int*)(ws + 0);
    int*      offs   = (int*)(ws + 64);
    int*      cursor = (int*)(ws + 128);
    int*      topi   = (int*)(ws + 256);                    // T*2 ints
    float*    gatew  = (float*)(ws + 65792);                // T*2 f32
    int*      tokl   = (int*)(ws + 131328);                 // PADCAP ints
    float*    wl     = (float*)(ws + 198912);               // PADCAP f32
    _Float16* hbuf   = (_Float16*)(ws + 266496);            // PADCAP*512 f16

    k_zero   <<<dim3((T_TOK * DDIM) / (256 * 4)), dim3(256), 0, stream>>>(out, counts);
    k_gate   <<<dim3(T_TOK / 8),   dim3(256), 0, stream>>>(x, Wg, topi, gatew, counts);
    k_scan   <<<dim3(1),           dim3(32),  0, stream>>>(counts, offs, cursor);
    k_scatter<<<dim3(T_TOK / 256), dim3(256), 0, stream>>>(topi, gatew, cursor, tokl, wl);
    k_pad    <<<dim3(NEXP),        dim3(64),  0, stream>>>(counts, offs, tokl, wl);

    const dim3 gg(MAXT64, NEXP, 1);
    k_fc1<<<gg, dim3(256), 0, stream>>>(x, W1, b1, counts, offs, tokl, hbuf);
    k_fc2<<<gg, dim3(256), 0, stream>>>(hbuf, W2, b2, gamma, beta, counts, offs, tokl, wl, out);
}
